// BiFormerLiteBlock_23124103922355
// MI455X (gfx1250) — matrix operand order repaired and checked by reading, not compiled, not run on hardware
//
#include <hip/hip_runtime.h>

// ---------------------------------------------------------------------------
// BiFormer-lite block for MI455X (gfx1250), wave32, WMMA bf16 (fp32 accum).
// Token-major bf16 activations; chained per-token linears composed on device.
// GEMM uses GLOBAL_LOAD_ASYNC_TO_LDS_B128 (ASYNCcnt) for LDS staging.
// ---------------------------------------------------------------------------

typedef __attribute__((ext_vector_type(16))) __bf16 v16bf;
typedef __attribute__((ext_vector_type(8)))  __bf16 v8bf;
typedef __attribute__((ext_vector_type(8)))  float  v8f;

#define NTOK   131072   // B*H*W = 8*128*128
#define HWSZ   16384
#define NBATCH 8

__device__ __forceinline__ v16bf cat16(v8bf lo, v8bf hi) {
  return __builtin_shufflevector(lo, hi, 0,1,2,3,4,5,6,7,8,9,10,11,12,13,14,15);
}
__device__ __forceinline__ v8f wmma_bf16(v16bf a, v16bf b, v8f c) {
  return __builtin_amdgcn_wmma_f32_16x16x32_bf16(false, a, false, b, (short)0, c, false, false);
}
__device__ __forceinline__ float gelu_f(float v) {
  return 0.5f * v * (1.0f + erff(v * 0.70710678118654752f));
}
// Async DMA: 16B per lane, global -> LDS (byte offset within wave LDS alloc).
__device__ __forceinline__ void async_g2l_b128(unsigned lds_off, const void* gptr) {
  asm volatile("global_load_async_to_lds_b128 %0, %1, off"
               :: "v"(lds_off), "v"(gptr) : "memory");
}
__device__ __forceinline__ void wait_async0() {
  asm volatile("s_wait_asynccnt 0x0" ::: "memory");
}

// ---------------------------------------------------------------------------
// WMMA GEMM:  out[M x Nout] = A[M x K] * Wt[Nout x K]^T + bias
// Block: 256 thr (8 waves), 128x128 tile, K chunked by 64 through 32KB
// dynamic LDS (As @ 0, Bs @ 16384) filled by async b128 loads.
// Epilogue: bias (+gelu) -> bf16 out, OR residual fp32 accumulate.
// ---------------------------------------------------------------------------
__global__ __launch_bounds__(256) void k_gemm(
    const __bf16* __restrict__ A, int lda,
    const __bf16* __restrict__ Wt, int K,
    const float* __restrict__ bias,
    __bf16* __restrict__ outB, int ldo,
    float* __restrict__ resid,
    int gelu_flag)
{
  extern __shared__ char smem[];                 // 32 KB dynamic LDS
  __bf16* As = (__bf16*)smem;                    // [128][64]
  __bf16* Bs = (__bf16*)(smem + 16384);          // [128][64]

  const int tid  = threadIdx.x;
  const int lane = tid & 31, wv = tid >> 5;
  const int row0 = blockIdx.x * 128, col0 = blockIdx.y * 128;
  const int g = lane >> 4, lr = lane & 15;

  v8f acc[8] = {};

  const int rs = tid >> 1, cb = (tid & 1) * 32;
  const __bf16* gA = A  + (size_t)(row0 + rs) * lda + cb;
  const __bf16* gB = Wt + (size_t)(col0 + rs) * K   + cb;
  const unsigned ldsA = (unsigned)(rs * 64 + cb) * 2u;
  const unsigned ldsB = 16384u + (unsigned)(rs * 64 + cb) * 2u;

  for (int kc = 0; kc < K; kc += 64) {
    __syncthreads();
#pragma unroll
    for (int u = 0; u < 4; ++u) {
      async_g2l_b128(ldsA + u * 16u, gA + kc + u * 8);
      async_g2l_b128(ldsB + u * 16u, gB + kc + u * 8);
    }
    wait_async0();
    __syncthreads();
#pragma unroll
    for (int ks = 0; ks < 2; ++ks) {
      const __bf16* pa = As + (wv * 16 + lr) * 64 + ks * 32 + g * 8;
      v16bf af = cat16(*(const v8bf*)pa, *(const v8bf*)(pa + 16));
      v16bf bfr[8];
#pragma unroll
      for (int nt = 0; nt < 8; ++nt) {
        const __bf16* pb = Bs + (nt * 16 + lr) * 64 + ks * 32 + g * 8;
        bfr[nt] = cat16(*(const v8bf*)pb, *(const v8bf*)(pb + 16));
      }
#pragma unroll
      for (int nt = 0; nt < 8; ++nt)
        acc[nt] = wmma_bf16(af, bfr[nt], acc[nt]);
    }
  }

#pragma unroll
  for (int nt = 0; nt < 8; ++nt) {
    const int col = col0 + nt * 16 + lr;
    const float bv = bias ? bias[col] : 0.0f;
#pragma unroll
    for (int e = 0; e < 8; ++e) {
      const int rr = row0 + wv * 16 + g * 8 + e;
      float v = acc[nt][e] + bv;
      if (gelu_flag) v = gelu_f(v);
      if (resid) resid[(size_t)rr * 256 + col] += v;
      else       outB[(size_t)rr * (size_t)ldo + col] = (__bf16)v;
    }
  }
}

// ---------------------------------------------------------------------------
// Local window attention: one 8x8 window per block, 4 waves x 2 heads.
// S = q k^T via 16 WMMAs (K=32 in one shot), shuffle softmax, O = S V.
// ---------------------------------------------------------------------------
__global__ __launch_bounds__(128) void k_local_attn(
    const __bf16* __restrict__ qh, const __bf16* __restrict__ kh,
    const __bf16* __restrict__ vh, __bf16* __restrict__ outb)
{
  __shared__ __bf16 Ss[4][64][64];
  __shared__ __bf16 vT[4][32][64];
  const int wi = blockIdx.x;
  const int b = wi >> 8, rem = wi & 255, wh = rem >> 4, ww = rem & 15;
  const int lane = threadIdx.x & 31, wv = threadIdx.x >> 5;
  const int g = lane >> 4, lr = lane & 15;

  auto tok = [&](int m) -> size_t {
    const int hh = wh * 8 + (m >> 3), www = ww * 8 + (m & 7);
    return (size_t)((b * 128 + hh) * 128 + www);
  };

  for (int hi = 0; hi < 2; ++hi) {
    const int head = wv * 2 + hi, co = head * 32;

    // stage V^T (per-wave, 32 channels x 64 tokens)
    for (int m = 0; m < 64; ++m)
      vT[wv][lane][m] = vh[tok(m) * 256 + co + lane];

    // S = q k^T  (scale pre-folded into q weights)
    v16bf af[4];
#pragma unroll
    for (int mt = 0; mt < 4; ++mt) {
      const __bf16* pq = qh + tok(mt * 16 + lr) * 256 + co + g * 8;
      af[mt] = cat16(*(const v8bf*)pq, *(const v8bf*)(pq + 16));
    }
    v8f sacc[4][4] = {};
#pragma unroll
    for (int nt = 0; nt < 4; ++nt) {
      const __bf16* pk = kh + tok(nt * 16 + lr) * 256 + co + g * 8;
      v16bf bf_ = cat16(*(const v8bf*)pk, *(const v8bf*)(pk + 16));
#pragma unroll
      for (int mt = 0; mt < 4; ++mt)
        sacc[mt][nt] = wmma_bf16(af[mt], bf_, sacc[mt][nt]);
    }

    // row softmax in D-fragment layout (rows split across 16-lane halves)
#pragma unroll
    for (int mt = 0; mt < 4; ++mt) {
#pragma unroll
      for (int r = 0; r < 8; ++r) {
        float x0 = sacc[mt][0][r], x1 = sacc[mt][1][r];
        float x2 = sacc[mt][2][r], x3 = sacc[mt][3][r];
        float mx = fmaxf(fmaxf(x0, x1), fmaxf(x2, x3));
        mx = fmaxf(mx, __shfl_xor(mx, 1));
        mx = fmaxf(mx, __shfl_xor(mx, 2));
        mx = fmaxf(mx, __shfl_xor(mx, 4));
        mx = fmaxf(mx, __shfl_xor(mx, 8));
        float e0 = __expf(x0 - mx), e1 = __expf(x1 - mx);
        float e2 = __expf(x2 - mx), e3 = __expf(x3 - mx);
        float sm = e0 + e1 + e2 + e3;
        sm += __shfl_xor(sm, 1); sm += __shfl_xor(sm, 2);
        sm += __shfl_xor(sm, 4); sm += __shfl_xor(sm, 8);
        const float inv = 1.0f / sm;
        const int m = mt * 16 + r + g * 8;
        Ss[wv][m][lr     ] = (__bf16)(e0 * inv);
        Ss[wv][m][lr + 16] = (__bf16)(e1 * inv);
        Ss[wv][m][lr + 32] = (__bf16)(e2 * inv);
        Ss[wv][m][lr + 48] = (__bf16)(e3 * inv);
      }
    }

    // O = S V
#pragma unroll
    for (int mt = 0; mt < 4; ++mt) {
      v8f oacc[2] = {};
#pragma unroll
      for (int kt = 0; kt < 2; ++kt) {
        const __bf16* ps = &Ss[wv][mt * 16 + lr][kt * 32 + g * 8];
        v16bf sa = cat16(*(const v8bf*)ps, *(const v8bf*)(ps + 16));
#pragma unroll
        for (int nt = 0; nt < 2; ++nt) {
          const __bf16* pv = &vT[wv][nt * 16 + lr][kt * 32 + g * 8];
          oacc[nt] = wmma_bf16(sa, cat16(*(const v8bf*)pv, *(const v8bf*)(pv + 16)), oacc[nt]);
        }
      }
#pragma unroll
      for (int nt = 0; nt < 2; ++nt)
#pragma unroll
        for (int e = 0; e < 8; ++e)
          outb[tok(mt * 16 + e + g * 8) * 256 + co + nt * 16 + lr] = (__bf16)oacc[nt][e];
    }
  }
}

// ---------------------------------------------------------------------------
// Global attention: grid = B*HEADS*128 blocks; block = 8 waves x 16 queries.
// K/V (64 tokens) staged once in LDS per block.
// ---------------------------------------------------------------------------
__global__ __launch_bounds__(256) void k_global_attn(
    const __bf16* __restrict__ qg, const __bf16* __restrict__ kg,
    const __bf16* __restrict__ vg, __bf16* __restrict__ outb)
{
  __shared__ __bf16 Ks[64][32];
  __shared__ __bf16 vT[32][64];
  __shared__ __bf16 Ss[8][16][64];
  const int blk = blockIdx.x;
  const int chunk = blk & 127, bh = blk >> 7, head = bh & 7, b = bh >> 3;
  const int tid = threadIdx.x, lane = tid & 31, wv = tid >> 5;
  const int g = lane >> 4, lr = lane & 15;
  const int co = head * 32;

  {
    const int j = tid >> 2, cbk = (tid & 3) * 8;
    *(v8bf*)&Ks[j][cbk] = *(const v8bf*)(kg + (size_t)(b * 64 + j) * 256 + co + cbk);
    const int c = tid >> 3, jb = (tid & 7) * 8;
    for (int u = 0; u < 8; ++u)
      vT[c][jb + u] = vg[(size_t)(b * 64 + jb + u) * 256 + co + c];
  }
  __syncthreads();

  const size_t t0 = (size_t)b * HWSZ + (size_t)chunk * 128 + (size_t)wv * 16;
  const __bf16* pq = qg + (t0 + lr) * 256 + co + g * 8;
  v16bf af = cat16(*(const v8bf*)pq, *(const v8bf*)(pq + 16));

  v8f sacc[4] = {};
#pragma unroll
  for (int nt = 0; nt < 4; ++nt) {
    const __bf16* pk = &Ks[nt * 16 + lr][g * 8];
    sacc[nt] = wmma_bf16(af, cat16(*(const v8bf*)pk, *(const v8bf*)(pk + 16)), sacc[nt]);
  }
#pragma unroll
  for (int r = 0; r < 8; ++r) {
    float x0 = sacc[0][r], x1 = sacc[1][r], x2 = sacc[2][r], x3 = sacc[3][r];
    float mx = fmaxf(fmaxf(x0, x1), fmaxf(x2, x3));
    mx = fmaxf(mx, __shfl_xor(mx, 1));
    mx = fmaxf(mx, __shfl_xor(mx, 2));
    mx = fmaxf(mx, __shfl_xor(mx, 4));
    mx = fmaxf(mx, __shfl_xor(mx, 8));
    float e0 = __expf(x0 - mx), e1 = __expf(x1 - mx);
    float e2 = __expf(x2 - mx), e3 = __expf(x3 - mx);
    float sm = e0 + e1 + e2 + e3;
    sm += __shfl_xor(sm, 1); sm += __shfl_xor(sm, 2);
    sm += __shfl_xor(sm, 4); sm += __shfl_xor(sm, 8);
    const float inv = 1.0f / sm;
    const int m = r + g * 8;
    Ss[wv][m][lr     ] = (__bf16)(e0 * inv);
    Ss[wv][m][lr + 16] = (__bf16)(e1 * inv);
    Ss[wv][m][lr + 32] = (__bf16)(e2 * inv);
    Ss[wv][m][lr + 48] = (__bf16)(e3 * inv);
  }

  v8f oacc[2] = {};
#pragma unroll
  for (int kt = 0; kt < 2; ++kt) {
    const __bf16* ps = &Ss[wv][lr][kt * 32 + g * 8];
    v16bf sa = cat16(*(const v8bf*)ps, *(const v8bf*)(ps + 16));
#pragma unroll
    for (int nt = 0; nt < 2; ++nt) {
      const __bf16* pv = &vT[nt * 16 + lr][kt * 32 + g * 8];
      oacc[nt] = wmma_bf16(sa, cat16(*(const v8bf*)pv, *(const v8bf*)(pv + 16)), oacc[nt]);
    }
  }
#pragma unroll
  for (int nt = 0; nt < 2; ++nt)
#pragma unroll
    for (int e = 0; e < 8; ++e)
      outb[(t0 + e + g * 8) * 256 + co + nt * 16 + lr] = (__bf16)oacc[nt][e];
}

// ---------------------------------------------------------------------------
// LayerNorm over C=256, token-major fp32 in -> bf16 out. One wave per token.
// ---------------------------------------------------------------------------
__global__ __launch_bounds__(256) void k_ln(
    const float* __restrict__ xin, const float* __restrict__ wgt,
    const float* __restrict__ bia, __bf16* __restrict__ outb)
{
  const int t = blockIdx.x * 8 + (threadIdx.x >> 5);
  const int lane = threadIdx.x & 31;
  const float* p = xin + (size_t)t * 256 + lane * 8;
  float v[8];
  ((float4*)v)[0] = ((const float4*)p)[0];
  ((float4*)v)[1] = ((const float4*)p)[1];
  float s = 0.0f, s2 = 0.0f;
#pragma unroll
  for (int i = 0; i < 8; ++i) { s += v[i]; s2 += v[i] * v[i]; }
  s  += __shfl_xor(s, 1);  s2 += __shfl_xor(s2, 1);
  s  += __shfl_xor(s, 2);  s2 += __shfl_xor(s2, 2);
  s  += __shfl_xor(s, 4);  s2 += __shfl_xor(s2, 4);
  s  += __shfl_xor(s, 8);  s2 += __shfl_xor(s2, 8);
  s  += __shfl_xor(s, 16); s2 += __shfl_xor(s2, 16);
  const float mean = s * (1.0f / 256.0f);
  const float var  = s2 * (1.0f / 256.0f) - mean * mean;
  const float inv  = rsqrtf(var + 1e-6f);
  v8bf ov;
#pragma unroll
  for (int i = 0; i < 8; ++i) {
    const float nv = (v[i] - mean) * inv;
    ov[i] = (__bf16)(wgt[lane * 8 + i] * nv + bia[lane * 8 + i]);
  }
  *(v8bf*)(outb + (size_t)t * 256 + lane * 8) = ov;
}

// ---------------------------------------------------------------------------
// BCHW <-> token-major transposes via LDS tiles (32c x 64p).
// ---------------------------------------------------------------------------
__global__ __launch_bounds__(256) void k_bchw_to_tok(
    const float* __restrict__ x, float* __restrict__ xr)
{
  __shared__ float tile[32][65];
  const int p0 = blockIdx.x * 64, c0 = blockIdx.y * 32, b = blockIdx.z;
  const int tid = threadIdx.x;
  {
    const int ci = tid >> 3, pb = (tid & 7) * 8;
    const float* gp = x + ((size_t)(b * 256 + c0 + ci)) * HWSZ + p0 + pb;
#pragma unroll
    for (int u = 0; u < 8; ++u) tile[ci][pb + u] = gp[u];
  }
  __syncthreads();
  {
    const int pi = tid >> 2, cbb = (tid & 3) * 8;
    float* op = xr + ((size_t)(b * HWSZ + p0 + pi)) * 256 + c0 + cbb;
#pragma unroll
    for (int u = 0; u < 8; ++u) op[u] = tile[cbb + u][pi];
  }
}

__global__ __launch_bounds__(256) void k_tok_to_bchw(
    const float* __restrict__ xr, float* __restrict__ out)
{
  __shared__ float tile[32][65];
  const int p0 = blockIdx.x * 64, c0 = blockIdx.y * 32, b = blockIdx.z;
  const int tid = threadIdx.x;
  {
    const int pi = tid >> 2, cbb = (tid & 3) * 8;
    const float* gp = xr + ((size_t)(b * HWSZ + p0 + pi)) * 256 + c0 + cbb;
#pragma unroll
    for (int u = 0; u < 8; ++u) tile[cbb + u][pi] = gp[u];
  }
  __syncthreads();
  {
    const int ci = tid >> 3, pb = (tid & 7) * 8;
    float* op = out + ((size_t)(b * 256 + c0 + ci)) * HWSZ + p0 + pb;
#pragma unroll
    for (int u = 0; u < 8; ++u) op[u] = tile[ci][pb + u];
  }
}

// ---------------------------------------------------------------------------
// score_map[t] = sum_c score[t][c]^2 ;  one wave per token.
// ---------------------------------------------------------------------------
__global__ __launch_bounds__(256) void k_smap(
    const __bf16* __restrict__ sc, float* __restrict__ smap)
{
  const int t = blockIdx.x * 8 + (threadIdx.x >> 5);
  const int lane = threadIdx.x & 31;
  v8bf v = *(const v8bf*)(sc + (size_t)t * 256 + lane * 8);
  float s = 0.0f;
#pragma unroll
  for (int i = 0; i < 8; ++i) { const float f = (float)v[i]; s += f * f; }
  s += __shfl_xor(s, 1); s += __shfl_xor(s, 2); s += __shfl_xor(s, 4);
  s += __shfl_xor(s, 8); s += __shfl_xor(s, 16);
  if (lane == 0) smap[t] = s;
}

// Iterative top-64 per batch (destructive on recomputed scratch).
__global__ __launch_bounds__(256) void k_topk(
    float* __restrict__ smap, int* __restrict__ idx)
{
  const int b = blockIdx.x, tid = threadIdx.x;
  float* s = smap + (size_t)b * HWSZ;
  __shared__ float bv[256];
  __shared__ int   bi[256];
  for (int k = 0; k < 64; ++k) {
    float best = -1e30f; int bidx = 0;
    for (int i = tid; i < HWSZ; i += 256) {
      const float v = s[i];
      if (v > best) { best = v; bidx = i; }
    }
    bv[tid] = best; bi[tid] = bidx;
    __syncthreads();
    for (int off = 128; off; off >>= 1) {
      if (tid < off && bv[tid + off] > bv[tid]) { bv[tid] = bv[tid + off]; bi[tid] = bi[tid + off]; }
      __syncthreads();
    }
    if (tid == 0) { idx[b * 64 + k] = bi[0]; s[bi[0]] = -1e30f; }
    __syncthreads();
  }
}

// Gather 64 tokens per batch (rows of 256 bf16 = 128 dwords).
__global__ __launch_bounds__(128) void k_gather(
    const __bf16* __restrict__ xn2, const int* __restrict__ idx,
    __bf16* __restrict__ xg)
{
  const int j = blockIdx.x;
  const int b = j >> 6;
  const size_t t = (size_t)b * HWSZ + (size_t)idx[j];
  ((float*)xg)[(size_t)j * 128 + threadIdx.x] =
      ((const float*)(xn2 + t * 256))[threadIdx.x];
}

// ---------------------------------------------------------------------------
// Weight prep: fp32 -> bf16 convert, and 256x256 composition W2@W1 (+bias).
// ---------------------------------------------------------------------------
__global__ __launch_bounds__(256) void k_convert(
    const float* __restrict__ w, const float* __restrict__ b,
    __bf16* __restrict__ wo, float* __restrict__ bo, int I, float scale)
{
  const int o = blockIdx.x;
  for (int i = threadIdx.x; i < I; i += blockDim.x)
    wo[(size_t)o * I + i] = (__bf16)(w[(size_t)o * I + i] * scale);
  if (threadIdx.x == 0) bo[o] = b[o] * scale;
}

__global__ __launch_bounds__(256) void k_compose(
    const float* __restrict__ W2, const float* __restrict__ b2,
    const float* __restrict__ W1, const float* __restrict__ b1,
    __bf16* __restrict__ Wo, float* __restrict__ bo, float scale)
{
  const int o = blockIdx.x, i = threadIdx.x;
  float acc = 0.0f;
  for (int c = 0; c < 256; ++c) acc += W2[o * 256 + c] * W1[c * 256 + i];
  Wo[(size_t)o * 256 + i] = (__bf16)(acc * scale);
  if (i == 0) {
    float bb = 0.0f;
    for (int c = 0; c < 256; ++c) bb += W2[o * 256 + c] * b1[c];
    bo[o] = (bb + b2[o]) * scale;
  }
}

// ---------------------------------------------------------------------------
extern "C" void kernel_launch(void* const* d_in, const int* in_sizes, int n_in,
                              void* d_out, int out_size, void* d_ws, size_t ws_size,
                              hipStream_t stream) {
  (void)in_sizes; (void)n_in; (void)out_size; (void)ws_size;
  const float* x       = (const float*)d_in[0];
  const float* n1w     = (const float*)d_in[1];
  const float* n1b     = (const float*)d_in[2];
  const float* qkv_w   = (const float*)d_in[3];
  const float* qkv_b   = (const float*)d_in[4];
  const float* lain_w  = (const float*)d_in[5];
  const float* lain_b  = (const float*)d_in[6];
  const float* laout_w = (const float*)d_in[7];
  const float* laout_b = (const float*)d_in[8];
  const float* projl_w = (const float*)d_in[9];
  const float* projl_b = (const float*)d_in[10];
  const float* n2w     = (const float*)d_in[11];
  const float* n2b     = (const float*)d_in[12];
  const float* qg_w    = (const float*)d_in[13];
  const float* qg_b    = (const float*)d_in[14];
  const float* kvg_w   = (const float*)d_in[15];
  const float* kvg_b   = (const float*)d_in[16];
  const float* gain_w  = (const float*)d_in[17];
  const float* gain_b  = (const float*)d_in[18];
  const float* gaout_w = (const float*)d_in[19];
  const float* gaout_b = (const float*)d_in[20];
  const float* projg_w = (const float*)d_in[21];
  const float* projg_b = (const float*)d_in[22];
  const float* n3w     = (const float*)d_in[23];
  const float* n3b     = (const float*)d_in[24];
  const float* ff1w    = (const float*)d_in[25];
  const float* ff1b    = (const float*)d_in[26];
  const float* ff2w    = (const float*)d_in[27];
  const float* ff2b    = (const float*)d_in[28];

  const float sc = 0.17677669529663688f;  // 1/sqrt(32)
  const size_t NT = NTOK;

  char* p = (char*)d_ws;
  auto alloc = [&](size_t bytes) -> void* {
    void* r = (void*)p;
    p += (bytes + 255) & ~(size_t)255;
    return r;
  };
  float*  xres = (float*) alloc(NT * 256 * 4);   // fp32 residual, token-major
  __bf16* xn   = (__bf16*)alloc(NT * 256 * 2);   // LN outputs
  __bf16* buf1 = (__bf16*)alloc(NT * 256 * 2);   // qh -> score -> hb(lo)
  __bf16* buf2 = (__bf16*)alloc(NT * 256 * 2);   // kh -> qh_glb -> hb(hi)
  __bf16* buf3 = (__bf16*)alloc(NT * 256 * 2);   // vh -> O_glb
  __bf16* buf4 = (__bf16*)alloc(NT * 256 * 2);   // O_local
  float*  smap = (float*) alloc(NT * 4);
  int*    idx  = (int*)   alloc(NBATCH * 64 * 4);
  __bf16* xg   = (__bf16*)alloc(512 * 256 * 2);
  __bf16* khg  = (__bf16*)alloc(512 * 256 * 2);
  __bf16* vhg  = (__bf16*)alloc(512 * 256 * 2);
  __bf16* Wqh  = (__bf16*)alloc(256*256*2); float* bqh  = (float*)alloc(256*4);
  __bf16* Wkh  = (__bf16*)alloc(256*256*2); float* bkh  = (float*)alloc(256*4);
  __bf16* Wvh  = (__bf16*)alloc(256*256*2); float* bvh  = (float*)alloc(256*4);
  __bf16* Wlo  = (__bf16*)alloc(256*256*2); float* blo  = (float*)alloc(256*4);
  __bf16* Wqg  = (__bf16*)alloc(256*256*2); float* bqg  = (float*)alloc(256*4);
  __bf16* Wgaq = (__bf16*)alloc(256*256*2); float* bgaq = (float*)alloc(256*4);
  __bf16* Wgk  = (__bf16*)alloc(256*256*2); float* bgk  = (float*)alloc(256*4);
  __bf16* Wgv  = (__bf16*)alloc(256*256*2); float* bgv  = (float*)alloc(256*4);
  __bf16* Wgo  = (__bf16*)alloc(256*256*2); float* bgo  = (float*)alloc(256*4);
  __bf16* Wf1  = (__bf16*)alloc(512*256*2); float* bf1  = (float*)alloc(512*4);
  __bf16* Wf2  = (__bf16*)alloc(256*512*2); float* bf2  = (float*)alloc(256*4);

  // ---- weight prep: compose chained per-token linears, convert to bf16 ----
  k_compose<<<256,256,0,stream>>>(lain_w,          lain_b,      qkv_w,         qkv_b,     Wqh, bqh, sc);
  k_compose<<<256,256,0,stream>>>(lain_w + 65536,  lain_b+256,  qkv_w + 65536, qkv_b+256, Wkh, bkh, 1.0f);
  k_compose<<<256,256,0,stream>>>(lain_w + 131072, lain_b+512,  qkv_w + 131072,qkv_b+512, Wvh, bvh, 1.0f);
  k_compose<<<256,256,0,stream>>>(projl_w, projl_b, laout_w, laout_b, Wlo, blo, 1.0f);
  k_compose<<<256,256,0,stream>>>(gain_w + 65536,  gain_b+256, kvg_w,         kvg_b,     Wgk, bgk, 1.0f);
  k_compose<<<256,256,0,stream>>>(gain_w + 131072, gain_b+512, kvg_w + 65536, kvg_b+256, Wgv, bgv, 1.0f);
  k_compose<<<256,256,0,stream>>>(projg_w, projg_b, gaout_w, gaout_b, Wgo, bgo, 1.0f);
  k_convert<<<256,256,0,stream>>>(qg_w,   qg_b,   Wqg,  bqg,  256, 1.0f);
  k_convert<<<256,256,0,stream>>>(gain_w, gain_b, Wgaq, bgaq, 256, sc);
  k_convert<<<512,256,0,stream>>>(ff1w,   ff1b,   Wf1,  bf1,  256, 1.0f);
  k_convert<<<256,256,0,stream>>>(ff2w,   ff2b,   Wf2,  bf2,  512, 1.0f);

  const dim3 g2(NT / 128, 2);
  const dim3 g4(NT / 128, 4);
  const int  smem = 32768;

  // ---- main pipeline ----
  k_bchw_to_tok<<<dim3(256, 8, 8), 256, 0, stream>>>(x, xres);
  k_ln<<<NT / 8, 256, 0, stream>>>(xres, n1w, n1b, xn);

  k_gemm<<<g2, 256, smem, stream>>>(xn, 256, Wqh, 256, bqh, buf1, 256, nullptr, 0);  // qh (scaled)
  k_gemm<<<g2, 256, smem, stream>>>(xn, 256, Wkh, 256, bkh, buf2, 256, nullptr, 0);  // kh
  k_gemm<<<g2, 256, smem, stream>>>(xn, 256, Wvh, 256, bvh, buf3, 256, nullptr, 0);  // vh
  k_local_attn<<<2048, 128, 0, stream>>>(buf1, buf2, buf3, buf4);
  k_gemm<<<g2, 256, smem, stream>>>(buf4, 256, Wlo, 256, blo, nullptr, 0, xres, 0);  // projl∘la_out + resid

  k_ln<<<NT / 8, 256, 0, stream>>>(xres, n2w, n2b, xn);
  k_gemm<<<g2, 256, smem, stream>>>(xn, 256, Wqg, 256, bqg, buf1, 256, nullptr, 0);  // score
  k_smap<<<NT / 8, 256, 0, stream>>>(buf1, smap);
  k_topk<<<8, 256, 0, stream>>>(smap, idx);
  k_gather<<<512, 128, 0, stream>>>(xn, idx, xg);
  k_gemm<<<dim3(4, 2), 256, smem, stream>>>(xg, 256, Wgk, 256, bgk, khg, 256, nullptr, 0);
  k_gemm<<<dim3(4, 2), 256, smem, stream>>>(xg, 256, Wgv, 256, bgv, vhg, 256, nullptr, 0);
  k_gemm<<<g2, 256, smem, stream>>>(buf1, 256, Wgaq, 256, bgaq, buf2, 256, nullptr, 0); // qh_glb
  k_global_attn<<<8192, 256, 0, stream>>>(buf2, khg, vhg, buf3);
  k_gemm<<<g2, 256, smem, stream>>>(buf3, 256, Wgo, 256, bgo, nullptr, 0, xres, 0);  // projg∘ga_out + resid

  k_ln<<<NT / 8, 256, 0, stream>>>(xres, n3w, n3b, xn);
  k_gemm<<<g4, 256, smem, stream>>>(xn, 256, Wf1, 256, bf1, buf1, 512, nullptr, 1);  // ffn1 + gelu -> [N,512]
  k_gemm<<<g2, 256, smem, stream>>>(buf1, 512, Wf2, 512, bf2, nullptr, 0, xres, 0);  // ffn2 + resid

  k_tok_to_bchw<<<dim3(256, 8, 8), 256, 0, stream>>>(xres, (float*)d_out);
}
